// BufferAttend1d_26130581029087
// MI455X (gfx1250) — matrix operand order repaired and checked by reading, not compiled, not run on hardware
//
#include <hip/hip_runtime.h>
#include <hip/hip_bf16.h>

// Problem constants (match reference)
#define BSZ    2
#define QLEN   2048
#define KLEN   2048
#define DMODEL 1024
#define NH     16
#define DHEAD  64
#define NPROJ  1024
#define MTOK   (BSZ * QLEN)
#define KW     (KLEN / 32)          // packed mask words per row

typedef __bf16 v8bf  __attribute__((ext_vector_type(8)));
typedef __bf16 v16bf __attribute__((ext_vector_type(16)));
typedef float  v8f   __attribute__((ext_vector_type(8)));
typedef unsigned int v4u __attribute__((ext_vector_type(4)));
typedef int v4i __attribute__((ext_vector_type(4)));
typedef int v8i __attribute__((ext_vector_type(8)));

union bf16x16u { v16bf v; v8bf h[2]; };

static __device__ __forceinline__ __bf16 f2bf(float f) {
    union { float f; unsigned u; } a; a.f = f;
    unsigned r = a.u + 0x7FFFu + ((a.u >> 16) & 1u);
    union { unsigned short s; __bf16 b; } o; o.s = (unsigned short)(r >> 16);
    return o.b;
}

// A-fragment (16x32 bf16): per lane, 8 bf16 at p and 8 bf16 at p+16 elements
static __device__ __forceinline__ v16bf load_afrag(const __bf16* p) {
    bf16x16u u;
    u.h[0] = *(const v8bf*)(p);
    u.h[1] = *(const v8bf*)(p + 16);
    return u.v;
}
// B-fragment (32x16 bf16): per lane, 8 bf16 at p and 8 bf16 at p+16 elements
static __device__ __forceinline__ v16bf load_bfrag(const __bf16* p) {
    bf16x16u u;
    u.h[0] = *(const v8bf*)(p);
    u.h[1] = *(const v8bf*)(p + 16);
    return u.v;
}

static __device__ __forceinline__ v8f wmma_bf16(v16bf a, v16bf b, v8f c) {
    return __builtin_amdgcn_wmma_f32_16x16x32_bf16(
        false, a, false, b, (short)0, c, false, false);
}

static __device__ __forceinline__ void wait_tensor0() {
#if __has_builtin(__builtin_amdgcn_s_wait_tensorcnt)
    __builtin_amdgcn_s_wait_tensorcnt(0);
#else
    asm volatile("s_wait_tensorcnt 0x0" ::: "memory");
#endif
}

// ---------------------------------------------------------------------------
// TDM: DMA a 2D bf16 tile (tile_d1 rows x tile_d0 elems, row stride stride0)
// from global memory into LDS, packed row-major. D# packing per ISA 8.3/8.4.
// ---------------------------------------------------------------------------
static __device__ __forceinline__ void tdm_load_2d(unsigned lds_addr, const void* gptr,
                                                   unsigned tensor_d0, unsigned tensor_d1,
                                                   unsigned tile_d0, unsigned tile_d1,
                                                   unsigned stride0)
{
    unsigned long long ga = (unsigned long long)(size_t)gptr;
    v4u g0;
    g0[0] = 1u;                                            // count=1, user descriptor
    g0[1] = lds_addr;                                      // LDS byte address
    g0[2] = (unsigned)(ga & 0xFFFFFFFFull);                // global addr [31:0]
    g0[3] = (unsigned)((ga >> 32) & 0x01FFFFFFull) | (2u << 30);  // [56:32] | type=2
    v8i g1;
    g1[0] = (int)(1u << 16);                               // wg_mask=0, data_size=2B
    g1[1] = (int)((tensor_d0 & 0xFFFFu) << 16);            // tensor_dim0[15:0]
    g1[2] = (int)(((tensor_d0 >> 16) & 0xFFFFu) | ((tensor_d1 & 0xFFFFu) << 16));
    g1[3] = (int)(((tensor_d1 >> 16) & 0xFFFFu) | ((tile_d0 & 0xFFFFu) << 16));
    g1[4] = (int)(tile_d1 & 0xFFFFu);                      // tile_dim1; tile_dim2=0
    g1[5] = (int)stride0;                                  // tensor_dim0_stride[31:0]
    g1[6] = 0;
    g1[7] = 0;
    v4i z4 = {0, 0, 0, 0};
#if __clang_major__ >= 23
    v8i z8 = {0, 0, 0, 0, 0, 0, 0, 0};
    __builtin_amdgcn_tensor_load_to_lds(g0, g1, z4, z4, z8, 0);
#else
    __builtin_amdgcn_tensor_load_to_lds(g0, g1, z4, z4, 0);
#endif
}

// ---------------------------------------------------------------------------
// Pack mask (int32 per key) into 1 bit per key with wave32 ballot.
// ---------------------------------------------------------------------------
__global__ void __launch_bounds__(256)
mask_pack_kernel(const int* __restrict__ mask, unsigned* __restrict__ packed)
{
    const int gid = blockIdx.x * 256 + threadIdx.x;
    const bool m = mask[gid] != 0;
    const unsigned bal = __builtin_amdgcn_ballot_w32(m);
    if ((threadIdx.x & 31) == 0) packed[gid >> 5] = bal;
}

// ---------------------------------------------------------------------------
// Projection GEMM: out(M x 1024) = in(M x 1024) @ W^T + bias, bf16 output.
//   mode 0/1 (q/k): ws[b][h][pos][d]   mode 2 (v): ws[b][h][d][pos]
// ---------------------------------------------------------------------------
__global__ void __launch_bounds__(256)
proj_bf16_kernel(const float* __restrict__ X, const float* __restrict__ W,
                 const float* __restrict__ bias, __bf16* __restrict__ out, int mode)
{
    __shared__ __align__(32) __bf16 As[128 * 32];
    __shared__ __align__(32) __bf16 Bs[64 * 32];

    const int tid  = threadIdx.x;
    const int wave = tid >> 5;
    const int lane = tid & 31;
    const int lrow = lane & 15;
    const int lhi  = lane >> 4;

    const int m0 = blockIdx.x * 128;
    const int n0 = blockIdx.y * 64;
    const int wm = (wave >> 1) * 32;
    const int wn = (wave & 1)  * 32;

    v8f acc[2][2] = {{{}, {}}, {{}, {}}};

    for (int k0 = 0; k0 < DMODEL; k0 += 32) {
        {
            const int r = tid >> 1;
            const int c = (tid & 1) * 16;
            const float* src = X + (size_t)(m0 + r) * DMODEL + k0 + c;
            __bf16* dst = As + r * 32 + c;
#pragma unroll
            for (int i = 0; i < 16; i += 4) {
                float4 f = *(const float4*)(src + i);
                dst[i + 0] = f2bf(f.x); dst[i + 1] = f2bf(f.y);
                dst[i + 2] = f2bf(f.z); dst[i + 3] = f2bf(f.w);
            }
        }
        {
            const int r = tid >> 2;
            const int c = (tid & 3) * 8;
            const float* src = W + (size_t)(n0 + r) * DMODEL + k0 + c;
            __bf16* dst = Bs + r * 32 + c;
#pragma unroll
            for (int i = 0; i < 8; i += 4) {
                float4 f = *(const float4*)(src + i);
                dst[i + 0] = f2bf(f.x); dst[i + 1] = f2bf(f.y);
                dst[i + 2] = f2bf(f.z); dst[i + 3] = f2bf(f.w);
            }
        }
        __syncthreads();

        v16bf afrag[2], bfrag[2];
#pragma unroll
        for (int mt = 0; mt < 2; ++mt)
            afrag[mt] = load_afrag(As + (wm + mt * 16 + lrow) * 32 + lhi * 8);
#pragma unroll
        for (int nt = 0; nt < 2; ++nt)
            bfrag[nt] = load_bfrag(Bs + (wn + nt * 16 + lrow) * 32 + lhi * 8);

#pragma unroll
        for (int mt = 0; mt < 2; ++mt)
#pragma unroll
            for (int nt = 0; nt < 2; ++nt)
                acc[mt][nt] = wmma_bf16(afrag[mt], bfrag[nt], acc[mt][nt]);

        __syncthreads();
    }

#pragma unroll
    for (int mt = 0; mt < 2; ++mt) {
#pragma unroll
        for (int nt = 0; nt < 2; ++nt) {
            const int ncol = n0 + wn + nt * 16 + lrow;
            const float bv = bias[ncol];
            const int h = ncol >> 6;
            const int d = ncol & 63;
#pragma unroll
            for (int j = 0; j < 8; ++j) {
                const int mrow = m0 + wm + mt * 16 + lhi * 8 + j;
                const int b    = mrow >> 11;
                const int pos  = mrow & 2047;
                const float val = acc[mt][nt][j] + bv;
                size_t idx;
                if (mode == 2)
                    idx = (((size_t)(b * NH + h) * DHEAD + d) * KLEN) + pos;
                else
                    idx = (((size_t)(b * NH + h) * QLEN + pos) * DHEAD) + d;
                out[idx] = f2bf(val);
            }
        }
    }
}

// ---------------------------------------------------------------------------
// Fused flash attention with TDM double-buffered K/V staging.
// Per wave: 16 q-rows x dv=64 for one (b,h). 8 waves/block -> 128 q rows.
// Wave 0 drives the Tensor Data Mover; TENSORcnt + block barrier sequence
// the double buffer. Mask consumed as packed bits (1 b64 load per row/block).
// ---------------------------------------------------------------------------
__global__ void __launch_bounds__(256)
attn_flash_kernel(const __bf16* __restrict__ qws, const __bf16* __restrict__ kws,
                  const __bf16* __restrict__ vws, const unsigned* __restrict__ pmask,
                  float* __restrict__ out)
{
    __shared__ __align__(32) __bf16 Ks[2][64 * 64];    // 2 x 8 KB key tiles
    __shared__ __align__(32) __bf16 Vs[2][64 * 64];    // 2 x 8 KB value tiles (d-major)
    __shared__ __align__(32) __bf16 plds[8][16][64];   // per-wave P staging, 16 KB

    const int tid  = threadIdx.x;
    const int wave = tid >> 5;
    const int lane = tid & 31;
    const int lrow = lane & 15;
    const int lhi  = lane >> 4;

    const int b  = blockIdx.z;
    const int h  = blockIdx.y;
    const int q0 = blockIdx.x * 128 + wave * 16;

    const __bf16* qbase = qws + (size_t)(b * NH + h) * QLEN * DHEAD;   // [q][d]
    const __bf16* kbase = kws + (size_t)(b * NH + h) * KLEN * DHEAD;   // [k][d]
    const __bf16* vbase = vws + (size_t)(b * NH + h) * DHEAD * KLEN;   // [d][k]
    const unsigned* pmb = pmask + ((size_t)b * QLEN + q0) * KW;        // [16 rows][KW]

    const unsigned ldsK0 = (unsigned)(size_t)&Ks[0][0];
    const unsigned ldsK1 = (unsigned)(size_t)&Ks[1][0];
    const unsigned ldsV0 = (unsigned)(size_t)&Vs[0][0];
    const unsigned ldsV1 = (unsigned)(size_t)&Vs[1][0];

    // Q A-fragments, resident across the key loop
    v16bf qfrag[2];
#pragma unroll
    for (int kk = 0; kk < 2; ++kk)
        qfrag[kk] = load_afrag(qbase + (size_t)(q0 + lrow) * DHEAD + kk * 32 + lhi * 8);

    v8f o[4] = {{}, {}, {}, {}};
    float mrun[8], lrun[8];
    unsigned valid = 0u;
#pragma unroll
    for (int j = 0; j < 8; ++j) { mrun[j] = -INFINITY; lrun[j] = 0.0f; }

    const float scale = 0.125f;   // 1/sqrt(64)

    // prime the pipeline: TDM pulls key-block 0 into buffer 0
    if (wave == 0) {
        tdm_load_2d(ldsK0, kbase, DHEAD, KLEN, 64, 64, DHEAD);
        tdm_load_2d(ldsV0, vbase, KLEN, DHEAD, 64, 64, KLEN);
    }

    for (int kb = 0; kb < KLEN; kb += 64) {
        const int buf = (kb >> 6) & 1;
        if (wave == 0) wait_tensor0();      // current buffer landed in LDS
        __syncthreads();                    // publish buffer; prev buffer now free
        if (wave == 0 && kb + 64 < KLEN) {  // start DMA of next block into other buffer
            const unsigned lk = buf ? ldsK0 : ldsK1;
            const unsigned lv = buf ? ldsV0 : ldsV1;
            tdm_load_2d(lk, kbase + (size_t)(kb + 64) * DHEAD, DHEAD, KLEN, 64, 64, DHEAD);
            tdm_load_2d(lv, vbase + (kb + 64),                 KLEN, DHEAD, 64, 64, KLEN);
        }

        // ---- S = scale * Q K^T from LDS key tile ----
        v8f s[4];
#pragma unroll
        for (int kt = 0; kt < 4; ++kt) {
            v8f c = {};
#pragma unroll
            for (int kk = 0; kk < 2; ++kk) {
                v16bf kf = load_bfrag(&Ks[buf][(kt * 16 + lrow) * 64 + kk * 32 + lhi * 8]);
                c = wmma_bf16(qfrag[kk], kf, c);
            }
            s[kt] = c;
        }

        // ---- mask via packed bits: one b64 word-pair per row covers 64 keys ----
#pragma unroll
        for (int j = 0; j < 8; ++j) {
            const int r = lhi * 8 + j;
            const unsigned long long wpair =
                *(const unsigned long long*)(pmb + (size_t)r * KW + (kb >> 5));
            valid |= (wpair != 0ull) ? (1u << r) : 0u;
#pragma unroll
            for (int kt = 0; kt < 4; ++kt) {
                const unsigned mk = (unsigned)(wpair >> (kt * 16 + lrow)) & 1u;
                const float v = s[kt][j] * scale;
                s[kt][j] = mk ? v : -4096.0f;
            }
        }

        // ---- online softmax (row ops across each 16-lane half) ----
        float mn[8];
#pragma unroll
        for (int j = 0; j < 8; ++j) {
            float v = fmaxf(fmaxf(s[0][j], s[1][j]), fmaxf(s[2][j], s[3][j]));
            v = fmaxf(v, __shfl_xor(v, 1, 32));
            v = fmaxf(v, __shfl_xor(v, 2, 32));
            v = fmaxf(v, __shfl_xor(v, 4, 32));
            v = fmaxf(v, __shfl_xor(v, 8, 32));
            mn[j] = fmaxf(mrun[j], v);
        }
#pragma unroll
        for (int j = 0; j < 8; ++j) {
            float rs = 0.0f;
#pragma unroll
            for (int kt = 0; kt < 4; ++kt) {
                const float p = __expf(s[kt][j] - mn[j]);
                s[kt][j] = p;
                rs += p;
            }
            rs += __shfl_xor(rs, 1, 32);
            rs += __shfl_xor(rs, 2, 32);
            rs += __shfl_xor(rs, 4, 32);
            rs += __shfl_xor(rs, 8, 32);
            const float alpha = __expf(mrun[j] - mn[j]);
            lrun[j] = lrun[j] * alpha + rs;
            mrun[j] = mn[j];
#pragma unroll
            for (int dt = 0; dt < 4; ++dt) o[dt][j] *= alpha;
#pragma unroll
            for (int kt = 0; kt < 4; ++kt)
                plds[wave][lhi * 8 + j][kt * 16 + lrow] = f2bf(s[kt][j]);
        }

        v16bf pfrag[2];
#pragma unroll
        for (int kk = 0; kk < 2; ++kk)
            pfrag[kk] = load_afrag(&plds[wave][lrow][kk * 32 + lhi * 8]);

        // ---- O += P V from LDS value tile ----
#pragma unroll
        for (int dt = 0; dt < 4; ++dt) {
#pragma unroll
            for (int kk = 0; kk < 2; ++kk) {
                v16bf vf = load_bfrag(&Vs[buf][(dt * 16 + lrow) * 64 + kk * 32 + lhi * 8]);
                o[dt] = wmma_bf16(pfrag[kk], vf, o[dt]);
            }
        }
    }

    // ---- epilogue: normalize, zero fully-masked rows, store fp32 ----
#pragma unroll
    for (int dt = 0; dt < 4; ++dt) {
        const int d = dt * 16 + lrow;
#pragma unroll
        for (int j = 0; j < 8; ++j) {
            const int r = lhi * 8 + j;
            const float val = ((valid >> r) & 1u) ? (o[dt][j] / lrun[j]) : 0.0f;
            out[((size_t)b * QLEN + q0 + r) * (NH * DHEAD) + h * DHEAD + d] = val;
        }
    }
}

// ---------------------------------------------------------------------------
extern "C" void kernel_launch(void* const* d_in, const int* in_sizes, int n_in,
                              void* d_out, int out_size, void* d_ws, size_t ws_size,
                              hipStream_t stream)
{
    (void)in_sizes; (void)n_in; (void)out_size; (void)ws_size;

    const float* x    = (const float*)d_in[0];
    const float* buf  = (const float*)d_in[1];
    const int*   mask = (const int*)d_in[2];
    const float* Wq   = (const float*)d_in[3];
    const float* bq   = (const float*)d_in[4];
    const float* Wk   = (const float*)d_in[5];
    const float* bk   = (const float*)d_in[6];
    const float* Wv   = (const float*)d_in[7];
    const float* bv   = (const float*)d_in[8];
    float* out = (float*)d_out;

    const size_t headElems = (size_t)BSZ * NH * QLEN * DHEAD;   // 4.19M bf16 each
    __bf16* qws = (__bf16*)d_ws;
    __bf16* kws = qws + headElems;
    __bf16* vws = kws + headElems;
    unsigned* pmask = (unsigned*)(vws + headElems);             // 1 MB packed mask

    mask_pack_kernel<<<(BSZ * QLEN * KLEN) / (256 * 32) * 32, 256, 0, stream>>>(mask, pmask);

    dim3 pgrid(MTOK / 128, NPROJ / 64);
    proj_bf16_kernel<<<pgrid, 256, 0, stream>>>(x,   Wq, bq, qws, 0);
    proj_bf16_kernel<<<pgrid, 256, 0, stream>>>(buf, Wk, bk, kws, 1);
    proj_bf16_kernel<<<pgrid, 256, 0, stream>>>(buf, Wv, bv, vws, 2);

    dim3 agrid(QLEN / 128, NH, BSZ);
    attn_flash_kernel<<<agrid, 256, 0, stream>>>(qws, kws, vws, pmask, out);
}
